// Q_Layer_17901423689857
// MI455X (gfx1250) — 8-bit matrix operand order repaired and checked by reading, compiled, not run on hardware
//
#include <hip/hip_runtime.h>
#include <stdint.h>
#include <stddef.h>


typedef int v8i __attribute__((ext_vector_type(8)));

#define DEV static __device__ __forceinline__

constexpr int TB = 64;
constexpr int NT = 196;
constexpr int DM = 768;
constexpr int HM = 3072;
constexpr int BN = TB * NT;
constexpr int BD = TB * DM;
constexpr int KA = 256;

DEV int clampi8(int v) { v = v < -128 ? -128 : v; return v > 127 ? 127 : v; }
DEV int rni(float f)   { return __float2int_rn(f); }

__global__ void absmax_kernel(const float* __restrict__ x, int n,
                              unsigned* __restrict__ out) {
  __shared__ float red[256];
  float m = 0.f;
  for (int i = blockIdx.x * blockDim.x + threadIdx.x; i < n;
       i += gridDim.x * blockDim.x)
    m = fmaxf(m, fabsf(x[i]));
  red[threadIdx.x] = m;
  __syncthreads();
  for (int s = 128; s; s >>= 1) {
    if ((int)threadIdx.x < s)
      red[threadIdx.x] = fmaxf(red[threadIdx.x], red[threadIdx.x + s]);
    __syncthreads();
  }
  if (!threadIdx.x) atomicMax(out, __float_as_uint(red[0]));
}

__global__ void quant_diag_kernel(const float* n1w, const float* n1b,
                                  const float* g1,  const float* n2w,
                                  const float* n2b, const float* g2,
                                  const float* sfs, const float* amax,
                                  int* n1i, int* n1bi, int* g1i,
                                  int* n2i, int* n2bi, int* g2i) {
  int d = blockIdx.x * blockDim.x + threadIdx.x;
  if (d >= DM) return;
  float ws0 = amax[0] * (1.f / 127.f);
  float ws2 = amax[2] * (1.f / 127.f);
  float ws3 = amax[3] * (1.f / 127.f);
  float ws6 = amax[6] * (1.f / 127.f);
  n1i[d]  = rni(n1w[d] / ws0);
  n1bi[d] = rni(n1b[d] / (sfs[0] * ws0));
  g1i[d]  = rni(g1[d]  / ws2);
  n2i[d]  = rni(n2w[d] / ws3);
  n2bi[d] = rni(n2b[d] / (sfs[3] * ws3));
  g2i[d]  = rni(g2[d]  / ws6);
}

__global__ void quant_w8_kernel(const float* __restrict__ w, int n,
                                int8_t* __restrict__ out,
                                const float* __restrict__ amax, int slot) {
  int i = blockIdx.x * blockDim.x + threadIdx.x;
  if (i >= n) return;
  float ws = amax[slot] * (1.f / 127.f);
  out[i] = (int8_t)clampi8(rni(w[i] / ws));
}

__global__ void quant_attn_w_kernel(const float* __restrict__ w,
                                    int8_t* __restrict__ out,
                                    const float* __restrict__ amax) {
  int i = blockIdx.x * blockDim.x + threadIdx.x;
  if (i >= KA * KA) return;
  int o = i >> 8, k = i & 255;
  int v = 0;
  if (o < NT && k < NT) {
    float ws = amax[1] * (1.f / 127.f);
    v = clampi8(rni(w[o * NT + k] / ws));
  }
  out[i] = (int8_t)v;
}

__global__ void quant_bias_kernel(const float* __restrict__ b, int n, int npad,
                                  int* __restrict__ out, const float* sfs,
                                  int sf_idx, const float* amax, int slot) {
  int i = blockIdx.x * blockDim.x + threadIdx.x;
  if (i >= npad) return;
  float ws = amax[slot] * (1.f / 127.f);
  out[i] = (i < n) ? rni(b[i] / (sfs[sf_idx] * ws)) : 0;
}

__global__ void prep_x_kernel(const float* __restrict__ x,
                              const float* __restrict__ sfs,
                              const float* __restrict__ amax,
                              const int* __restrict__ n1i,
                              const int* __restrict__ n1bi,
                              int8_t* __restrict__ x0q,
                              int8_t* __restrict__ x1q) {
  int idx = blockIdx.x * blockDim.x + threadIdx.x;
  if (idx >= TB * NT * DM) return;
  int d = idx % DM;
  int r = idx / DM;
  int n = r % NT;
  int b = r / NT;
  float a0 = sfs[0];
  float r1 = a0 * (amax[0] * (1.f / 127.f)) / sfs[1];
  int x0 = clampi8(rni(x[idx] / a0));
  x0q[idx] = (int8_t)x0;
  int t  = x0 * n1i[d] + n1bi[d];
  int y1 = clampi8(rni((float)t * r1));
  x1q[((size_t)(b * DM + d)) * KA + n] = (int8_t)y1;
}

struct EpiP {
  const int*    bias;
  const int8_t* idq;
  const int*    g1i;
  const int*    n2i;
  const int*    n2bi;
  int8_t*       out8a;
  int8_t*       out8b;
  float*        outf;
  const float*  sfs;
  const float*  amax;
};

template <int LDA, int LDB, int KSTEPS, int EPI>
__global__ __launch_bounds__(256) void gemm_i8_kernel(
    const int8_t* __restrict__ A, const int8_t* __restrict__ Bq, EpiP P) {
  const int lane = threadIdx.x & 31;
  const int wave = threadIdx.x >> 5;
  const int wm   = wave & 1;
  const int wn   = wave >> 1;
  const int m0   = blockIdx.y * 128 + wm * 64;
  const int n0   = blockIdx.x * 128 + wn * 32;
  const int half = lane >> 4;
  const int lr   = lane & 15;

  v8i acc[4][2];
#pragma unroll
  for (int i = 0; i < 4; i++)
#pragma unroll
    for (int j = 0; j < 2; j++) {
      v8i z = {0, 0, 0, 0, 0, 0, 0, 0};
      acc[i][j] = z;
    }

  for (int ks = 0; ks < KSTEPS; ++ks) {
    const int k0 = ks * 64;
    v8i a[4], bfr[2];
#pragma unroll
    for (int i = 0; i < 4; i++) {
      const int8_t* ap = A + (size_t)(m0 + i * 16 + lr) * LDA + k0 + half * 16;
      int4 t0 = *(const int4*)(ap);
      int4 t1 = *(const int4*)(ap + 32);
      a[i][0] = t0.x; a[i][1] = t0.y; a[i][2] = t0.z; a[i][3] = t0.w;
      a[i][4] = t1.x; a[i][5] = t1.y; a[i][6] = t1.z; a[i][7] = t1.w;
      if (ks + 1 < KSTEPS) __builtin_prefetch(ap + 64, 0, 0);
    }
#pragma unroll
    for (int j = 0; j < 2; j++) {
      const int8_t* bp = Bq + (size_t)(n0 + j * 16 + lr) * LDB + k0 + half * 16;
      int4 q0 = *(const int4*)(bp);
      int4 q1 = *(const int4*)(bp + 32);
      bfr[j][0] = q0.x; bfr[j][1] = q0.y; bfr[j][2] = q0.z; bfr[j][3] = q0.w;
      bfr[j][4] = q1.x; bfr[j][5] = q1.y; bfr[j][6] = q1.z; bfr[j][7] = q1.w;
      if (ks + 1 < KSTEPS) __builtin_prefetch(bp + 64, 0, 0);
    }
#pragma unroll
    for (int i = 0; i < 4; i++)
#pragma unroll
      for (int j = 0; j < 2; j++)
        acc[i][j] = __builtin_amdgcn_wmma_i32_16x16x64_iu8(
            true, a[i], true, bfr[j], acc[i][j], false, false);
  }

  if constexpr (EPI == 0) {
    const float r2  = P.sfs[1] * (P.amax[1] * (1.f / 127.f)) / P.sfs[2];
    const float r3  = P.sfs[2] * (P.amax[2] * (1.f / 127.f)) / P.sfs[3];
    const float rid = P.sfs[0] / P.sfs[3];
    const float r4  = P.sfs[3] * (P.amax[3] * (1.f / 127.f)) / P.sfs[4];
#pragma unroll
    for (int i = 0; i < 4; i++) {
      const int mt    = m0 + i * 16;
      const int bb    = mt / DM;
      const int dbase = mt - bb * DM;
      const int d0    = dbase + half * 8;
#pragma unroll
      for (int j = 0; j < 2; j++) {
        const int n = n0 + j * 16 + lr;
        if (n >= NT) continue;
        const size_t rowoff = ((size_t)(bb * NT + n)) * DM + d0;
        const int8_t* idp = P.idq + rowoff;
        const int bia = P.bias[n];
        unsigned lo3 = 0, hi3 = 0, lo4 = 0, hi4 = 0;
#pragma unroll
        for (int v = 0; v < 8; v++) {
          const int d  = d0 + v;
          int s  = acc[i][j][v] + bia;
          int y2 = clampi8(rni((float)s * r2));
          int t  = y2 * P.g1i[d];
          int y3 = clampi8(rni((float)t * r3) + rni((float)idp[v] * rid));
          int t2 = y3 * P.n2i[d] + P.n2bi[d];
          int y4 = clampi8(rni((float)t2 * r4));
          const unsigned sh = (unsigned)((v & 3) * 8);
          if (v < 4) { lo3 |= (unsigned)(y3 & 255) << sh;
                       lo4 |= (unsigned)(y4 & 255) << sh; }
          else       { hi3 |= (unsigned)(y3 & 255) << sh;
                       hi4 |= (unsigned)(y4 & 255) << sh; }
        }
        *(int2*)(P.out8a + rowoff) = make_int2((int)lo3, (int)hi3);
        *(int2*)(P.out8b + rowoff) = make_int2((int)lo4, (int)hi4);
      }
    }
  } else if constexpr (EPI == 1) {
    const float r5 = P.sfs[4] * (P.amax[4] * (1.f / 127.f)) / P.sfs[5];
#pragma unroll
    for (int i = 0; i < 4; i++) {
      const int mb = m0 + i * 16 + half * 8;
#pragma unroll
      for (int j = 0; j < 2; j++) {
        const int o   = n0 + j * 16 + lr;
        const int bia = P.bias[o];
#pragma unroll
        for (int v = 0; v < 8; v++) {
          int s = acc[i][j][v] + bia;
          s = s < 0 ? 0 : s;
          P.out8a[(size_t)(mb + v) * HM + o] = (int8_t)clampi8(rni((float)s * r5));
        }
      }
    }
  } else {
    const float r6  = P.sfs[5] * (P.amax[5] * (1.f / 127.f)) / P.sfs[6];
    const float r7  = P.sfs[6] * (P.amax[6] * (1.f / 127.f)) / P.sfs[7];
    const float rid = P.sfs[3] / P.sfs[7];
    const float a7  = P.sfs[7];
#pragma unroll
    for (int i = 0; i < 4; i++) {
      const int mb = m0 + i * 16 + half * 8;
#pragma unroll
      for (int j = 0; j < 2; j++) {
        const int o   = n0 + j * 16 + lr;
        const int bia = P.bias[o];
        const int g2  = P.g1i[o];
#pragma unroll
        for (int v = 0; v < 8; v++) {
          const int m = mb + v;
          int s  = acc[i][j][v] + bia;
          int y6 = clampi8(rni((float)s * r6));
          int t  = y6 * g2;
          int y7 = clampi8(rni((float)t * r7) +
                           rni((float)P.idq[(size_t)m * DM + o] * rid));
          P.outf[(size_t)m * DM + o] = (float)y7 * a7;
        }
      }
    }
  }
}

__global__ void tail_kernel(float* out, int out_size, const float* sfs) {
  if (threadIdx.x == 0 && blockIdx.x == 0) out[out_size - 1] = sfs[7];
}

extern "C" void kernel_launch(void* const* d_in, const int* in_sizes, int n_in,
                              void* d_out, int out_size, void* d_ws,
                              size_t ws_size, hipStream_t stream) {
  (void)in_sizes; (void)n_in; (void)ws_size;
  const float* x    = (const float*)d_in[0];
  const float* n1w  = (const float*)d_in[1];
  const float* n1b  = (const float*)d_in[2];
  const float* attw = (const float*)d_in[3];
  const float* attb = (const float*)d_in[4];
  const float* g1   = (const float*)d_in[5];
  const float* n2w  = (const float*)d_in[6];
  const float* n2b  = (const float*)d_in[7];
  const float* fc1w = (const float*)d_in[8];
  const float* fc1b = (const float*)d_in[9];
  const float* fc2w = (const float*)d_in[10];
  const float* fc2b = (const float*)d_in[11];
  const float* g2   = (const float*)d_in[12];
  const float* sfs  = (const float*)d_in[13];

  char* ws = (char*)d_ws;
  size_t off = 0;
  auto take = [&](size_t bytes) -> char* {
    off = (off + 255) & ~(size_t)255;
    char* p = ws + off;
    off += bytes;
    return p;
  };

  float*  amax  = (float*)take(7 * sizeof(float));
  int*    n1i   = (int*)take(DM * 4);
  int*    n1bi  = (int*)take(DM * 4);
  int*    g1i   = (int*)take(DM * 4);
  int*    n2i   = (int*)take(DM * 4);
  int*    n2bi  = (int*)take(DM * 4);
  int*    g2i   = (int*)take(DM * 4);
  int*    attbi = (int*)take(KA * 4);
  int*    fc1bi = (int*)take(HM * 4);
  int*    fc2bi = (int*)take(DM * 4);
  int8_t* attwq = (int8_t*)take((size_t)KA * KA);
  int8_t* fc1wq = (int8_t*)take((size_t)HM * DM);
  int8_t* fc2wq = (int8_t*)take((size_t)DM * HM);
  int8_t* x0q   = (int8_t*)take((size_t)TB * NT * DM);
  int8_t* x1q   = (int8_t*)take((size_t)BD * KA);
  int8_t* y3q   = (int8_t*)take((size_t)TB * NT * DM);
  int8_t* y4q   = (int8_t*)take((size_t)TB * NT * DM);
  int8_t* y5q   = (int8_t*)take((size_t)BN * HM);

  hipMemsetAsync(amax, 0, 7 * sizeof(float), stream);
  hipMemsetAsync(x1q, 0, (size_t)BD * KA, stream);

  auto blks = [](int n) { int b = (n + 255) / 256; return b > 1024 ? 1024 : b; };
  absmax_kernel<<<blks(DM), 256, 0, stream>>>(n1w, DM, (unsigned*)(amax + 0));
  absmax_kernel<<<blks(NT * NT), 256, 0, stream>>>(attw, NT * NT, (unsigned*)(amax + 1));
  absmax_kernel<<<blks(DM), 256, 0, stream>>>(g1, DM, (unsigned*)(amax + 2));
  absmax_kernel<<<blks(DM), 256, 0, stream>>>(n2w, DM, (unsigned*)(amax + 3));
  absmax_kernel<<<blks(HM * DM), 256, 0, stream>>>(fc1w, HM * DM, (unsigned*)(amax + 4));
  absmax_kernel<<<blks(DM * HM), 256, 0, stream>>>(fc2w, DM * HM, (unsigned*)(amax + 5));
  absmax_kernel<<<blks(DM), 256, 0, stream>>>(g2, DM, (unsigned*)(amax + 6));

  quant_diag_kernel<<<3, 256, 0, stream>>>(n1w, n1b, g1, n2w, n2b, g2, sfs, amax,
                                           n1i, n1bi, g1i, n2i, n2bi, g2i);
  quant_attn_w_kernel<<<(KA * KA) / 256, 256, 0, stream>>>(attw, attwq, amax);
  quant_w8_kernel<<<(HM * DM) / 256, 256, 0, stream>>>(fc1w, HM * DM, fc1wq, amax, 4);
  quant_w8_kernel<<<(DM * HM) / 256, 256, 0, stream>>>(fc2w, DM * HM, fc2wq, amax, 5);
  quant_bias_kernel<<<1, 256, 0, stream>>>(attb, NT, KA, attbi, sfs, 1, amax, 1);
  quant_bias_kernel<<<HM / 256, 256, 0, stream>>>(fc1b, HM, HM, fc1bi, sfs, 4, amax, 4);
  quant_bias_kernel<<<DM / 256, 256, 0, stream>>>(fc2b, DM, DM, fc2bi, sfs, 5, amax, 5);

  prep_x_kernel<<<(TB * NT * DM) / 256, 256, 0, stream>>>(x, sfs, amax, n1i, n1bi,
                                                          x0q, x1q);

  {
    EpiP P{};
    P.bias = attbi; P.idq = x0q; P.g1i = g1i; P.n2i = n2i; P.n2bi = n2bi;
    P.out8a = y3q; P.out8b = y4q; P.sfs = sfs; P.amax = amax;
    gemm_i8_kernel<KA, KA, 4, 0><<<dim3(2, BD / 128), 256, 0, stream>>>(x1q, attwq, P);
  }
  {
    EpiP P{};
    P.bias = fc1bi; P.out8a = y5q; P.sfs = sfs; P.amax = amax;
    gemm_i8_kernel<DM, DM, 12, 1><<<dim3(HM / 128, BN / 128), 256, 0, stream>>>(y4q, fc1wq, P);
  }
  {
    EpiP P{};
    P.bias = fc2bi; P.idq = y3q; P.g1i = g2i; P.outf = (float*)d_out;
    P.sfs = sfs; P.amax = amax;
    gemm_i8_kernel<HM, HM, 48, 2><<<dim3(DM / 128, BN / 128), 256, 0, stream>>>(y5q, fc2wq, P);
  }
  tail_kernel<<<1, 32, 0, stream>>>((float*)d_out, out_size, sfs);
}
